// DeepGMM_13941463843225
// MI455X (gfx1250) — matrix operand order repaired and checked by reading, not compiled, not run on hardware
//
#include <hip/hip_runtime.h>
#include <hip/hip_bf16.h>
#include <math.h>

// ---------------------------------------------------------------- constants
#define NB    8192
#define YD    512
#define XD    64
#define KC    16
#define SS    10
#define NS    (NB*SS)          // 81920
#define LOG2PI 1.8378770664093453f

typedef _Float16 v16h __attribute__((ext_vector_type(16)));
typedef _Float16 h8   __attribute__((ext_vector_type(8)));
typedef float    v8f  __attribute__((ext_vector_type(8)));

// ---------------------------------------------------------------- helpers
__device__ __forceinline__ float softplusf(float v) {
    return v > 20.f ? v : log1pf(expf(v));
}
__device__ __forceinline__ float wred32(float v) {       // full wave32 sum
#pragma unroll
    for (int m = 16; m >= 1; m >>= 1) v += __shfl_xor(v, m, 32);
    return v;
}
__device__ __forceinline__ float wred16s(float v) {      // sum within 16-lane group
#pragma unroll
    for (int m = 8; m >= 1; m >>= 1) v += __shfl_xor(v, m, 32);
    return v;
}
__device__ __forceinline__ float wred16m(float v) {      // max within 16-lane group
#pragma unroll
    for (int m = 8; m >= 1; m >>= 1) v = fmaxf(v, __shfl_xor(v, m, 32));
    return v;
}

// A tile (16x32 f16, row-major source, row stride in halfs).
// lane<16: row = m0+lane,    K halves = [k0+0..7 , k0+8..15]
// lane>=16: row = m0+lane-16, K halves = [k0+16..23, k0+24..31]
__device__ __forceinline__ v16h load_a16(const _Float16* base, int stride,
                                         int m0, int k0, int lane) {
    int row = m0 + (lane & 15);
    int h   = lane >> 4;
    const h8* p0 = (const h8*)(base + (size_t)row * stride + k0 + h * 16);
    const h8* p1 = (const h8*)(base + (size_t)row * stride + k0 + h * 16 + 8);
    h8 x0 = *p0, x1 = *p1;
    v16h a;
#pragma unroll
    for (int i = 0; i < 8; ++i) { a[i] = x0[i]; a[8 + i] = x1[i]; }
    return a;
}

// B tile (32x16 f16) from a PRE-TRANSPOSED weight WT[N][K] (row-major, row
// stride = Kdim halfs). Lane needs column n0+(lane&15), K = kb..kb+15 which is
// 16 CONTIGUOUS halfs -> two b128 loads.
__device__ __forceinline__ v16h load_bt16(const _Float16* WT, int Kdim,
                                          int k0, int n0, int lane) {
    int col = n0 + (lane & 15);
    int kb  = k0 + (lane >> 4) * 16;
    const h8* p = (const h8*)(WT + (size_t)col * Kdim + kb);
    h8 x0 = p[0], x1 = p[1];
    v16h b;
#pragma unroll
    for (int i = 0; i < 8; ++i) { b[i] = x0[i]; b[8 + i] = x1[i]; }
    return b;
}

__device__ __forceinline__ v8f wmma_f16(v16h a, v16h b, v8f c) {
    return __builtin_amdgcn_wmma_f32_16x16x32_f16(false, a, false, b,
                                                  (short)0, c, false, false);
}

// ---------------------------------------------------------------- kernel 0: prep
// Convert Y to f16; build TRANSPOSED f16 weight copies; zero accumulators.
__global__ __launch_bounds__(256) void prep_kernel(
    const float* __restrict__ Y, const float* __restrict__ We_mu,
    const float* __restrict__ We_sig, const float* __restrict__ Wd_mu,
    const float* __restrict__ Wd_sig,
    _Float16* __restrict__ Yh, _Float16* __restrict__ WeMuT,
    _Float16* __restrict__ WeSigT, _Float16* __restrict__ WdMuT,
    _Float16* __restrict__ WdSigT, float* __restrict__ acc) {
    const int SY = NB * YD, SW = YD * XD;
    int gid = blockIdx.x * blockDim.x + threadIdx.x;
    if (gid < 4) acc[gid] = 0.f;
    if (gid < SY) { Yh[gid] = (_Float16)Y[gid]; return; }
    int g = gid - SY;
    if (g < SW) {               // WeMuT[n*512+k] = We_mu[k*64+n]
        int n = g >> 9, k = g & 511;
        WeMuT[g] = (_Float16)We_mu[k * XD + n];
        return;
    }
    g -= SW;
    if (g < SW) {
        int n = g >> 9, k = g & 511;
        WeSigT[g] = (_Float16)We_sig[k * XD + n];
        return;
    }
    g -= SW;
    if (g < SW) {               // WdMuT[n*64+k] = Wd_mu[k*512+n]
        int n = g >> 6, k = g & 63;
        WdMuT[g] = (_Float16)Wd_mu[k * YD + n];
        return;
    }
    g -= SW;
    if (g < SW) {
        int n = g >> 6, k = g & 63;
        WdSigT[g] = (_Float16)Wd_sig[k * YD + n];
    }
}

// ---------------------------------------------------------------- kernel 1: encoder GEMM
// One wave = one 16x16 tile of enc_mu or enc_sig. tiles = 512(M) * 4(N) * 2.
__global__ __launch_bounds__(256) void enc_gemm_kernel(
    const _Float16* __restrict__ Yh, const _Float16* __restrict__ WeMuT,
    const _Float16* __restrict__ WeSigT, const float* __restrict__ be_mu,
    const float* __restrict__ be_sig, float* __restrict__ enc_mu,
    float* __restrict__ enc_sig) {
    int wave = threadIdx.x >> 5, lane = threadIdx.x & 31;
    int t = blockIdx.x * 8 + wave;
    int which = t & 1, nt = (t >> 1) & 3, mt = t >> 3;
    int m0 = mt * 16, n0 = nt * 16;
    const _Float16* WT = which ? WeSigT : WeMuT;
    v8f c;
#pragma unroll
    for (int i = 0; i < 8; ++i) c[i] = 0.f;
#pragma unroll 4
    for (int k0 = 0; k0 < YD; k0 += 32) {
        v16h a = load_a16(Yh, YD, m0, k0, lane);
        v16h b = load_bt16(WT, YD, k0, n0, lane);
        c = wmma_f16(a, b, c);
    }
    float* out = which ? enc_sig : enc_mu;
    const float* bias = which ? be_sig : be_mu;
    int col = n0 + (lane & 15);
    float bv = bias[col];
#pragma unroll
    for (int r = 0; r < 8; ++r) {
        int row = m0 + (lane >> 4) * 8 + r;
        float v = c[r] + bv;
        if (which) v = softplusf(v) + 1e-3f;
        out[(size_t)row * XD + col] = v;
    }
}

// ---------------------------------------------------------------- kernel 2: GMM per-row
// One wave per row n. Lane owns dims (lane, lane+32). k = lane&15 duplication trick.
__global__ __launch_bounds__(256) void cluster_kernel(
    const float* __restrict__ enc_mu, const float* __restrict__ enc_sig,
    const float* __restrict__ phi_mus, const float* __restrict__ phi_sigs,
    const float* __restrict__ phi_logits, float* __restrict__ zlp,
    float* __restrict__ mu_t, float* __restrict__ Sig_t,
    float* __restrict__ acc) {
    int wave = threadIdx.x >> 5, lane = threadIdx.x & 31;
    int n = blockIdx.x * 8 + wave;
    int kl = lane & 15;
    // log_softmax(phi_logits) for this lane's k
    float pl = phi_logits[kl];
    float pm_ = wred16m(pl);
    float pse = wred16s(expf(pl - pm_));
    float lp  = pl - pm_ - logf(pse);

    int d0 = lane, d1 = lane + 32;
    float emu0 = enc_mu[(size_t)n * XD + d0], emu1 = enc_mu[(size_t)n * XD + d1];
    float es0  = enc_sig[(size_t)n * XD + d0], es1 = enc_sig[(size_t)n * XD + d1];
    float ie0 = 1.f / es0, ie1 = 1.f / es1;
    float mylogit = 0.f;
#pragma unroll
    for (int k = 0; k < KC; ++k) {
        float pmu0 = phi_mus[k * XD + d0], pmu1 = phi_mus[k * XD + d1];
        float psg0 = phi_sigs[k * XD + d0], psg1 = phi_sigs[k * XD + d1];
        float st0 = es0 + psg0, st1 = es1 + psg1;
        float q0 = (emu0 - pmu0) / st0, q1 = (emu1 - pmu1) / st1;
        float q = q0 * q0 + q1 * q1;
        float h = logf(st0) + logf(st1);
        float ip0 = 1.f / psg0, ip1 = 1.f / psg1;
        float Sg0 = 1.f / (ie0 + ip0), Sg1 = 1.f / (ie1 + ip1);
        size_t base = ((size_t)n * KC + k) * XD;
        Sig_t[base + d0] = Sg0;
        Sig_t[base + d1] = Sg1;
        mu_t[base + d0] = Sg0 * (ie0 * emu0 + ip0 * pmu0);
        mu_t[base + d1] = Sg1 * (ie1 * emu1 + ip1 * pmu1);
        float qs = wred32(q);
        float hs = wred32(h);
        float logit = -0.5f * ((float)XD * LOG2PI + qs) - hs;
        if (kl == k) mylogit = logit;
    }
    mylogit += lp;
    // log_softmax over K
    float mm = wred16m(mylogit);
    float ss = wred16s(expf(mylogit - mm));
    float lsm = mylogit - mm - logf(ss);
    if (lane < 16) zlp[(size_t)n * KC + lane] = lsm;
    // loss5 = log(sum_k exp(z_log_probs))
    float se2 = wred16s(expf(lsm));
    if (lane == 0) atomicAdd(acc + 2, logf(se2));
}

// ---------------------------------------------------------------- kernel 3: sample + losses 2/3/4
// One wave per (n,s) pair.
__global__ __launch_bounds__(256) void sample_kernel(
    const float* __restrict__ zlp, const float* __restrict__ mu_t,
    const float* __restrict__ Sig_t, const float* __restrict__ enc_mu,
    const float* __restrict__ enc_sig, const float* __restrict__ theta_mus,
    const float* __restrict__ theta_sigs, const float* __restrict__ theta_logits,
    const float* __restrict__ phi_mus, const float* __restrict__ phi_sigs,
    const float* __restrict__ u_noise, const float* __restrict__ eps_noise,
    const float* __restrict__ temperature, _Float16* __restrict__ xh,
    float* __restrict__ acc) {
    int wave = threadIdx.x >> 5, lane = threadIdx.x & 31;
    int pair = blockIdx.x * 8 + wave;
    int n = pair / SS;
    int kl = lane & 15;

    float l = zlp[(size_t)n * KC + kl];
    float u = u_noise[(size_t)pair * KC + kl];
    float g = -logf(-logf(u));
    float T = temperature[0];
    float aa = (l + g) / T;
    float mm = wred16m(aa);
    float ex = expf(aa - mm);
    float se = wred16s(ex);
    float z  = ex / se;                                  // lane's cluster weight

    float tl = theta_logits[kl];
    float tm = wred16m(tl);
    float tse = wred16s(expf(tl - tm));
    float tlp = tl - tm - logf(tse);
    float l3b = wred16s(tlp * z);                        // sum log_softmax(theta)*z
    float l4b = wred16s(l * z);                          // sum z_log_probs*z

    int d0 = lane, d1 = lane + 32;
    float mu0 = 0.f, mu1 = 0.f, Sg0 = 0.f, Sg1 = 0.f;
    float tM0 = 0.f, tM1 = 0.f, tS0 = 0.f, tS1 = 0.f;
    float pM0 = 0.f, pM1 = 0.f, pS0 = 0.f, pS1 = 0.f;
#pragma unroll
    for (int k = 0; k < KC; ++k) {
        float zk = __shfl(z, k, 32);
        size_t base = ((size_t)n * KC + k) * XD;
        mu0 += zk * mu_t[base + d0];  mu1 += zk * mu_t[base + d1];
        Sg0 += zk * Sig_t[base + d0]; Sg1 += zk * Sig_t[base + d1];
        tM0 += zk * theta_mus[k * XD + d0];  tM1 += zk * theta_mus[k * XD + d1];
        tS0 += zk * theta_sigs[k * XD + d0]; tS1 += zk * theta_sigs[k * XD + d1];
        pM0 += zk * phi_mus[k * XD + d0];    pM1 += zk * phi_mus[k * XD + d1];
        pS0 += zk * phi_sigs[k * XD + d0];   pS1 += zk * phi_sigs[k * XD + d1];
    }
    float e0 = eps_noise[(size_t)pair * XD + d0];
    float e1 = eps_noise[(size_t)pair * XD + d1];
    float x0 = mu0 + sqrtf(Sg0) * e0;
    float x1 = mu1 + sqrtf(Sg1) * e1;
    xh[(size_t)pair * XD + d0] = (_Float16)x0;
    xh[(size_t)pair * XD + d1] = (_Float16)x1;

    float emu0 = enc_mu[(size_t)n * XD + d0], emu1 = enc_mu[(size_t)n * XD + d1];
    float es0  = enc_sig[(size_t)n * XD + d0], es1 = enc_sig[(size_t)n * XD + d1];

    float a2_0 = (x0 - emu0) / es0, a2_1 = (x1 - emu1) / es1;
    float a3_0 = (x0 - tM0) / tS0,  a3_1 = (x1 - tM1) / tS1;
    float a4_0 = (x0 - pM0) / pS0,  a4_1 = (x1 - pM1) / pS1;
    float v = 0.5f * (a2_0 * a2_0 + a2_1 * a2_1) + logf(es0) + logf(es1)   // loss2 var
            - 0.5f * (a3_0 * a3_0 + a3_1 * a3_1) - logf(tS0) - logf(tS1)   // loss3 var
            + 0.5f * (a4_0 * a4_0 + a4_1 * a4_1) + logf(pS0) + logf(pS1);  // loss4 var
    float vs = wred32(v);
    if (lane == 0) atomicAdd(acc + 1, vs + l3b - l4b);
}

// ---------------------------------------------------------------- kernel 4: decoder GEMM + loss1
// One wave = one 16x32 slab of BOTH mu_y and sig_y (4 accumulators, 8 WMMAs).
// tiles = 5120(M) * 16(N32). A tile loaded once, reused for 2 column tiles x 2 mats.
__global__ __launch_bounds__(256) void dec_gemm_kernel(
    const _Float16* __restrict__ xh, const _Float16* __restrict__ WdMuT,
    const _Float16* __restrict__ WdSigT, const float* __restrict__ bd_mu,
    const float* __restrict__ bd_sig, const float* __restrict__ Y,
    float* __restrict__ acc) {
    int wave = threadIdx.x >> 5, lane = threadIdx.x & 31;
    int tile = blockIdx.x * 8 + wave;
    int mt = tile >> 4, nt = tile & 15;
    int m0 = mt * 16, n0 = nt * 32;
    v8f cm0, cm1, cs0, cs1;
#pragma unroll
    for (int i = 0; i < 8; ++i) { cm0[i] = 0.f; cm1[i] = 0.f; cs0[i] = 0.f; cs1[i] = 0.f; }
#pragma unroll
    for (int k0 = 0; k0 < XD; k0 += 32) {
        v16h a   = load_a16(xh, XD, m0, k0, lane);
        v16h bm0 = load_bt16(WdMuT,  XD, k0, n0,      lane);
        v16h bm1 = load_bt16(WdMuT,  XD, k0, n0 + 16, lane);
        v16h bs0 = load_bt16(WdSigT, XD, k0, n0,      lane);
        v16h bs1 = load_bt16(WdSigT, XD, k0, n0 + 16, lane);
        cm0 = wmma_f16(a, bm0, cm0);
        cm1 = wmma_f16(a, bm1, cm1);
        cs0 = wmma_f16(a, bs0, cs0);
        cs1 = wmma_f16(a, bs1, cs1);
    }
    float t = 0.f;
#pragma unroll
    for (int j = 0; j < 2; ++j) {
        int col = n0 + j * 16 + (lane & 15);
        float bmv = bd_mu[col], bsv = bd_sig[col];
        const v8f& cm = j ? cm1 : cm0;
        const v8f& cs = j ? cs1 : cs0;
#pragma unroll
        for (int r = 0; r < 8; ++r) {
            int row = m0 + (lane >> 4) * 8 + r;
            int norig = row / SS;                  // broadcast Y over samples
            float y  = Y[(size_t)norig * YD + col];
            float mu = cm[r] + bmv;
            float sg = softplusf(cs[r] + bsv) + 1e-3f;
            float dd = (y - mu) / sg;
            t += -0.5f * dd * dd - logf(sg);
        }
    }
    float ts = wred32(t);
    if (lane == 0) atomicAdd(acc + 0, ts);
}

// ---------------------------------------------------------------- kernel 5: finalize
__global__ void finalize_kernel(const float* __restrict__ acc,
                                float* __restrict__ out) {
    // net analytic constant per (n,s) row across loss1..loss4: -224*log(2*pi)
    float cst = (float)NS * (-224.0f * LOG2PI);
    float tot = (acc[0] + acc[1] + cst) / (float)SS + acc[2];
    out[0] = -tot;
}

// ---------------------------------------------------------------- launch
extern "C" void kernel_launch(void* const* d_in, const int* in_sizes, int n_in,
                              void* d_out, int out_size, void* d_ws, size_t ws_size,
                              hipStream_t stream) {
    (void)in_sizes; (void)n_in; (void)out_size; (void)ws_size;
    const float* Y            = (const float*)d_in[0];
    const float* We_mu        = (const float*)d_in[1];
    const float* be_mu        = (const float*)d_in[2];
    const float* We_sig       = (const float*)d_in[3];
    const float* be_sig       = (const float*)d_in[4];
    const float* Wd_mu        = (const float*)d_in[5];
    const float* bd_mu        = (const float*)d_in[6];
    const float* Wd_sig       = (const float*)d_in[7];
    const float* bd_sig       = (const float*)d_in[8];
    const float* phi_mus      = (const float*)d_in[9];
    const float* phi_sigs     = (const float*)d_in[10];
    const float* phi_logits   = (const float*)d_in[11];
    const float* theta_mus    = (const float*)d_in[12];
    const float* theta_sigs   = (const float*)d_in[13];
    const float* theta_logits = (const float*)d_in[14];
    const float* u_noise      = (const float*)d_in[15];
    const float* eps_noise    = (const float*)d_in[16];
    const float* temperature  = (const float*)d_in[17];
    float* out = (float*)d_out;

    // workspace layout (256B aligned)
    char* w = (char*)d_ws;
    size_t off = 0;
    auto take = [&](size_t bytes) {
        char* p = w + off;
        off = (off + bytes + 255) & ~(size_t)255;
        return p;
    };
    _Float16* Yh     = (_Float16*)take((size_t)NB * YD * 2);
    _Float16* WeMuT  = (_Float16*)take((size_t)YD * XD * 2);
    _Float16* WeSigT = (_Float16*)take((size_t)YD * XD * 2);
    _Float16* WdMuT  = (_Float16*)take((size_t)XD * YD * 2);
    _Float16* WdSigT = (_Float16*)take((size_t)XD * YD * 2);
    float* enc_mu    = (float*)take((size_t)NB * XD * 4);
    float* enc_sig   = (float*)take((size_t)NB * XD * 4);
    float* zlp       = (float*)take((size_t)NB * KC * 4);
    float* mu_t      = (float*)take((size_t)NB * KC * XD * 4);
    float* Sig_t     = (float*)take((size_t)NB * KC * XD * 4);
    _Float16* xh     = (_Float16*)take((size_t)NS * XD * 2);
    float* acc       = (float*)take(4 * 4);

    // 0) convert + transpose weights + zero accumulators
    {
        int total = NB * YD + 4 * YD * XD;
        int blocks = (total + 255) / 256;
        prep_kernel<<<blocks, 256, 0, stream>>>(Y, We_mu, We_sig, Wd_mu, Wd_sig,
                                                Yh, WeMuT, WeSigT, WdMuT, WdSigT, acc);
    }
    // 1) encoder GEMM (512 Mtiles * 4 Ntiles * 2 matrices = 4096 wave-tiles)
    enc_gemm_kernel<<<512, 256, 0, stream>>>(Yh, WeMuT, WeSigT, be_mu, be_sig,
                                             enc_mu, enc_sig);
    // 2) per-row GMM responsibilities + posterior
    cluster_kernel<<<NB / 8, 256, 0, stream>>>(enc_mu, enc_sig, phi_mus, phi_sigs,
                                               phi_logits, zlp, mu_t, Sig_t, acc);
    // 3) gumbel-softmax, einsums, reparam sample, loss2/3/4
    sample_kernel<<<NS / 8, 256, 0, stream>>>(zlp, mu_t, Sig_t, enc_mu, enc_sig,
                                              theta_mus, theta_sigs, theta_logits,
                                              phi_mus, phi_sigs, u_noise, eps_noise,
                                              temperature, xh, acc);
    // 4) decoder GEMM fused with loss1 (5120 Mtiles * 16 N32-slabs = 81920 waves)
    dec_gemm_kernel<<<(5120 * 16) / 8, 256, 0, stream>>>(xh, WdMuT, WdSigT,
                                                         bd_mu, bd_sig, Y, acc);
    // 5) finalize scalar
    finalize_kernel<<<1, 1, 0, stream>>>(acc, out);
}
